// ActorGraphPolicy_80796924772663
// MI455X (gfx1250) — matrix operand order repaired and checked by reading, not compiled, not run on hardware
//
#include <hip/hip_runtime.h>

#define LSEQ 20
#define SDIM 64
#define BTOT 16384

typedef __attribute__((ext_vector_type(16))) _Float16 v16h;
typedef __attribute__((ext_vector_type(8)))  _Float16 v8h;
typedef __attribute__((ext_vector_type(8)))  float    v8f;

__device__ __forceinline__ void lds_fence() {
  // per-wave LDS ordering + compiler memory fence (DScnt covers the LDS
  // portion even if accesses lower to flat ops)
  asm volatile("s_wait_dscnt 0" ::: "memory");
}

// ---------------------------------------------------------------------------
// Prepack: f32 row-major W[K][N] -> f16 WMMA B-fragment tiles.
// CDNA5 16-bit B 32x16 layout: lane holds col n = lane%16; halves h=0..7 hold
// rows k = h + 8*(lane/16), h=8..15 hold k = 16 + 8*(lane/16) + (h-8). Tiles ordered [kt][nt][lane][h].
// ---------------------------------------------------------------------------
__global__ void prepack_w(const float* __restrict__ W, _Float16* __restrict__ dst,
                          int K, int N, int Npad) {
  int ntiles = Npad >> 4;
  int total  = (K >> 5) * ntiles * 512;
  int idx = blockIdx.x * blockDim.x + threadIdx.x;
  if (idx >= total) return;
  int h    = idx & 15;
  int lane = (idx >> 4) & 31;
  int tile = idx >> 9;
  int nt = tile % ntiles;
  int kt = tile / ntiles;
  int k = kt * 32 + ((lane >> 4) << 3) + (h & 7) + ((h >> 3) << 4);
  int n = nt * 16 + (lane & 15);
  float v = (n < N) ? W[k * N + n] : 0.0f;
  dst[idx] = (_Float16)v;
}

// ---------------------------------------------------------------------------
// 32-row GEMM: out[32 x NTILES*16] = op(act[32 x KK] @ Wpacked + bias)
// Two 16-row A-fragments share every B-fragment (halves B-side L2/WGP$
// traffic per FLOP, gives ILP=2 on the WMMA accumulator chains).
// A-fragment (16-bit 16x32): lane m=lane%16; halves 0-7 -> k0..k0+7,
// halves 8-15 -> k0+16..k0+23, k0 = kt*32 + 8*(lane/16).
// OP: 0 = store, 1 = relu, 2 = tanh.
// ---------------------------------------------------------------------------
template<int KK, int NTILES, int OP>
__device__ __forceinline__ void gemm32(const _Float16* act,
                                       const _Float16* __restrict__ wpk,
                                       const float* __restrict__ bias,
                                       _Float16* outb, int lane) {
  const int m  = lane & 15;
  const int hi = lane >> 4;
  #pragma unroll
  for (int nt = 0; nt < NTILES; ++nt) {
    v8f c0 = {}, c1 = {};
    #pragma unroll
    for (int kt = 0; kt < KK / 32; ++kt) {
      const int k0 = kt * 32 + hi * 8;
      v8h a0lo = *(const v8h*)(act + m * KK + k0);
      v8h a0hi = *(const v8h*)(act + m * KK + k0 + 16);
      v8h a1lo = *(const v8h*)(act + (m + 16) * KK + k0);
      v8h a1hi = *(const v8h*)(act + (m + 16) * KK + k0 + 16);
      v16h a0, a1;
      #pragma unroll
      for (int i = 0; i < 8; ++i) {
        a0[i] = a0lo[i]; a0[i + 8] = a0hi[i];
        a1[i] = a1lo[i]; a1[i + 8] = a1hi[i];
      }
      v16h b = *(const v16h*)(wpk + (size_t)(((kt * NTILES + nt) << 5) + lane) * 16);
      c0 = __builtin_amdgcn_wmma_f32_16x16x32_f16(false, a0, false, b,
                                                  (short)0, c0, false, false);
      c1 = __builtin_amdgcn_wmma_f32_16x16x32_f16(false, a1, false, b,
                                                  (short)0, c1, false, false);
    }
    const int n  = nt * 16 + m;
    const float bv = bias[n];
    #pragma unroll
    for (int r = 0; r < 8; ++r) {
      float v0 = c0[r] + bv;
      float v1 = c1[r] + bv;
      if (OP == 1) { v0 = fmaxf(v0, 0.0f); v1 = fmaxf(v1, 0.0f); }
      if (OP == 2) { v0 = tanhf(v0);       v1 = tanhf(v1); }
      outb[(r + hi * 8)      * (NTILES * 16) + n] = (_Float16)v0;
      outb[(16 + r + hi * 8) * (NTILES * 16) + n] = (_Float16)v1;
    }
  }
  lds_fence();
}

// Row-wise L2 normalize of a 32x64 f16 LDS tile (one row per lane).
__device__ __forceinline__ void normalize32x64(_Float16* buf, float* nrm, int lane) {
  lds_fence();
  {
    float s = 0.0f;
    #pragma unroll
    for (int j = 0; j < 64; ++j) { float v = (float)buf[lane * 64 + j]; s += v * v; }
    nrm[lane] = 1.0f / fmaxf(sqrtf(s), 1e-12f);
  }
  lds_fence();
  #pragma unroll
  for (int i = 0; i < 64; ++i) {
    int e = i * 32 + lane;
    buf[e] = (_Float16)((float)buf[e] * nrm[e >> 6]);
  }
  lds_fence();
}

// ---------------------------------------------------------------------------
// Bottom-up scan (t = L-1 .. 0): msg recurrence, spill msgs (f16) to global.
// One wave per block, 32 batch rows per wave.
// ---------------------------------------------------------------------------
__global__ void __launch_bounds__(32) bu_kernel(
    const float* __restrict__ state,
    const _Float16* __restrict__ wU1, const _Float16* __restrict__ wU2,
    const _Float16* __restrict__ wU3,
    const float* __restrict__ ub1, const float* __restrict__ ub2,
    const float* __restrict__ ub3,
    _Float16* __restrict__ msgs) {
  __shared__ _Float16 xm  [4096];   // 32 x 128
  __shared__ _Float16 bufA[8192];   // 32 x 256 (only 32x64 used here)
  __shared__ _Float16 bufB[8192];
  __shared__ _Float16 mbuf[2048];   // 32 x 64
  __shared__ float    nrm [32];
  const int lane = threadIdx.x & 31;
  const int b0   = blockIdx.x * 32;

  #pragma unroll
  for (int i = 0; i < 64; ++i) mbuf[i * 32 + lane] = (_Float16)0.0f;
  lds_fence();

  for (int t = LSEQ - 1; t >= 0; --t) {
    // x tile (32x64) -> bufB
    #pragma unroll
    for (int i = 0; i < 64; ++i) {
      int e = i * 32 + lane; int row = e >> 6, col = e & 63;
      bufB[e] = (_Float16)state[(b0 + row) * (LSEQ * SDIM) + t * SDIM + col];
    }
    lds_fence();
    gemm32<64, 4, 0>(bufB, wU1, ub1, bufA, lane);    // h_pre = x@uW1+b1
    normalize32x64(bufA, nrm, lane);                 // h
    #pragma unroll
    for (int i = 0; i < 128; ++i) {                  // xm = tanh([h, m])
      int e = i * 32 + lane; int row = e >> 7, col = e & 127;
      float v = (col < 64) ? (float)bufA[row * 64 + col]
                           : (float)mbuf[row * 64 + (col - 64)];
      xm[e] = (_Float16)tanhf(v);
    }
    lds_fence();
    gemm32<128, 4, 2>(xm,   wU2, ub2, bufB, lane);   // h2 = tanh(xm@uW2+b2)
    gemm32<64,  4, 0>(bufB, wU3, ub3, mbuf, lane);   // msg_pre
    normalize32x64(mbuf, nrm, lane);                 // msg
    #pragma unroll
    for (int i = 0; i < 64; ++i) {
      int e = i * 32 + lane; int row = e >> 6, col = e & 63;
      msgs[((size_t)(b0 + row) * LSEQ + t) * 64 + col] = mbuf[e];
    }
    lds_fence();
  }
}

// ---------------------------------------------------------------------------
// Top-down scan (t = 0 .. L-1): action + message MLPs, write actions (B,L).
// ---------------------------------------------------------------------------
__global__ void __launch_bounds__(32) td_kernel(
    const _Float16* __restrict__ msgs,
    const _Float16* __restrict__ wA1, const _Float16* __restrict__ wA2,
    const _Float16* __restrict__ wA3,
    const _Float16* __restrict__ wM1, const _Float16* __restrict__ wM2,
    const _Float16* __restrict__ wM3,
    const float* __restrict__ ab1, const float* __restrict__ ab2,
    const float* __restrict__ ab3,
    const float* __restrict__ mb1, const float* __restrict__ mb2,
    const float* __restrict__ mb3,
    float* __restrict__ out) {
  __shared__ _Float16 xm  [4096];   // 32 x 128
  __shared__ _Float16 bufA[8192];   // 32 x 256
  __shared__ _Float16 bufB[8192];   // 32 x 256
  __shared__ _Float16 mbuf[2048];   // 32 x 64
  __shared__ float    nrm [32];
  const int lane = threadIdx.x & 31;
  const int b0   = blockIdx.x * 32;
  const int m    = lane & 15, hi = lane >> 4;

  #pragma unroll
  for (int i = 0; i < 64; ++i) mbuf[i * 32 + lane] = (_Float16)0.0f;
  lds_fence();
  const float ab3_0 = ab3[0];

  for (int t = 0; t < LSEQ; ++t) {
    // xm = tanh([mu_t, m])
    #pragma unroll
    for (int i = 0; i < 128; ++i) {
      int e = i * 32 + lane; int row = e >> 7, col = e & 127;
      float v = (col < 64)
          ? (float)msgs[((size_t)(b0 + row) * LSEQ + t) * 64 + col]
          : (float)mbuf[row * 64 + (col - 64)];
      xm[e] = (_Float16)tanhf(v);
    }
    lds_fence();

    // action MLP: 128->256 relu, 256->256 relu, 256->1 tanh
    gemm32<128, 16, 1>(xm,   wA1, ab1, bufA, lane);
    gemm32<256, 16, 1>(bufA, wA2, ab2, bufB, lane);
    {
      v8f c0 = {}, c1 = {};
      #pragma unroll
      for (int kt = 0; kt < 8; ++kt) {
        const int k0 = kt * 32 + hi * 8;
        v8h a0lo = *(const v8h*)(bufB + m * 256 + k0);
        v8h a0hi = *(const v8h*)(bufB + m * 256 + k0 + 16);
        v8h a1lo = *(const v8h*)(bufB + (m + 16) * 256 + k0);
        v8h a1hi = *(const v8h*)(bufB + (m + 16) * 256 + k0 + 16);
        v16h a0, a1;
        #pragma unroll
        for (int i = 0; i < 8; ++i) {
          a0[i] = a0lo[i]; a0[i + 8] = a0hi[i];
          a1[i] = a1lo[i]; a1[i + 8] = a1hi[i];
        }
        v16h b = *(const v16h*)(wA3 + (size_t)((kt << 5) + lane) * 16);
        c0 = __builtin_amdgcn_wmma_f32_16x16x32_f16(false, a0, false, b,
                                                    (short)0, c0, false, false);
        c1 = __builtin_amdgcn_wmma_f32_16x16x32_f16(false, a1, false, b,
                                                    (short)0, c1, false, false);
      }
      if (m == 0) {                       // only column 0 is the real output
        #pragma unroll
        for (int r = 0; r < 8; ++r) {
          out[(size_t)(b0 + r + hi * 8)      * LSEQ + t] = tanhf(c0[r] + ab3_0);
          out[(size_t)(b0 + 16 + r + hi * 8) * LSEQ + t] = tanhf(c1[r] + ab3_0);
        }
      }
    }

    // message MLP: 128->256 relu, 256->256 relu, 256->64, normalize
    gemm32<128, 16, 1>(xm,   wM1, mb1, bufA, lane);
    gemm32<256, 16, 1>(bufA, wM2, mb2, bufB, lane);
    gemm32<256,  4, 0>(bufB, wM3, mb3, mbuf, lane);
    normalize32x64(mbuf, nrm, lane);
  }
}

// ---------------------------------------------------------------------------
extern "C" void kernel_launch(void* const* d_in, const int* in_sizes, int n_in,
                              void* d_out, int out_size, void* d_ws, size_t ws_size,
                              hipStream_t stream) {
  (void)in_sizes; (void)n_in; (void)out_size; (void)ws_size;
  const float* state = (const float*)d_in[0];
  const float* uW1 = (const float*)d_in[1];  const float* ub1 = (const float*)d_in[2];
  const float* uW2 = (const float*)d_in[3];  const float* ub2 = (const float*)d_in[4];
  const float* uW3 = (const float*)d_in[5];  const float* ub3 = (const float*)d_in[6];
  const float* aW1 = (const float*)d_in[7];  const float* ab1 = (const float*)d_in[8];
  const float* aW2 = (const float*)d_in[9];  const float* ab2 = (const float*)d_in[10];
  const float* aW3 = (const float*)d_in[11]; const float* ab3 = (const float*)d_in[12];
  const float* mW1 = (const float*)d_in[13]; const float* mb1 = (const float*)d_in[14];
  const float* mW2 = (const float*)d_in[15]; const float* mb2 = (const float*)d_in[16];
  const float* mW3 = (const float*)d_in[17]; const float* mb3 = (const float*)d_in[18];

  char* ws = (char*)d_ws;
  const size_t MSG_BYTES = (size_t)BTOT * LSEQ * 64 * sizeof(_Float16); // ~42 MB
  _Float16* msgs = (_Float16*)ws;
  _Float16* wpk  = (_Float16*)(ws + MSG_BYTES);

  // packed-weight offsets in halfs
  const size_t OU1 = 0,      OU2 = 4096,   OU3 = 12288;
  const size_t OA1 = 16384,  OA2 = 49152,  OA3 = 114688;
  const size_t OM1 = 118784, OM2 = 151552, OM3 = 217088;

  struct PD { const float* W; size_t off; int K, N, Npad; };
  const PD pd[9] = {
    {uW1, OU1,  64,  64,  64}, {uW2, OU2, 128,  64,  64}, {uW3, OU3,  64,  64,  64},
    {aW1, OA1, 128, 256, 256}, {aW2, OA2, 256, 256, 256}, {aW3, OA3, 256,   1,  16},
    {mW1, OM1, 128, 256, 256}, {mW2, OM2, 256, 256, 256}, {mW3, OM3, 256,  64,  64},
  };
  for (int i = 0; i < 9; ++i) {
    int total = (pd[i].K >> 5) * (pd[i].Npad >> 4) * 512;
    prepack_w<<<(total + 255) / 256, 256, 0, stream>>>(pd[i].W, wpk + pd[i].off,
                                                       pd[i].K, pd[i].N, pd[i].Npad);
  }

  dim3 grid(BTOT / 32), block(32);   // 1 wave per block, 32 rows per wave
  bu_kernel<<<grid, block, 0, stream>>>(state, wpk + OU1, wpk + OU2, wpk + OU3,
                                        ub1, ub2, ub3, msgs);
  td_kernel<<<grid, block, 0, stream>>>(msgs, wpk + OA1, wpk + OA2, wpk + OA3,
                                        wpk + OM1, wpk + OM2, wpk + OM3,
                                        ab1, ab2, ab3, mb1, mb2, mb3,
                                        (float*)d_out);
}
